// Model_80711025426508
// MI455X (gfx1250) — matrix operand order repaired and checked by reading, not compiled, not run on hardware
//
#include <hip/hip_runtime.h>
#include <hip/hip_bf16.h>

// ---------------------------------------------------------------------------
// Problem constants (match reference)
// ---------------------------------------------------------------------------
#define BB   8
#define NV   16
#define HH   192      // token axis
#define DD   512
#define NHD  8        // heads
#define HD   64       // head dim
#define DFF  2048
#define TOK  (BB*NV*HH)       // 24576 token rows
#define NSL  (BB*NV)          // 128 (b,n) slices
#define NSLICE (NSL*NHD)      // 1024 attention slices
#define EPS  1e-5f

typedef __attribute__((ext_vector_type(16))) _Float16 v16h;
typedef __attribute__((ext_vector_type(8)))  _Float16 h8;
typedef __attribute__((ext_vector_type(8)))  float    v8f;
typedef __attribute__((ext_vector_type(4)))  unsigned int u32x4;
typedef __attribute__((ext_vector_type(8)))  int      i32x8;
typedef __attribute__((ext_vector_type(4)))  int      i32x4;

// ---------------------------------------------------------------------------
// WMMA fragment helpers (wave32, 16x16x32 f16 -> f32)
// ---------------------------------------------------------------------------
__device__ inline v16h frag_a_rm(const _Float16* p, int ld) {
  // A[m][k] = p[m*ld + k]
  const int lane = threadIdx.x & 31;
  const int m  = lane & 15;
  const int kb = (lane & 16) ? 16 : 0;
  const _Float16* r = p + (size_t)m * ld;
  v16h a;
#pragma unroll
  for (int j = 0; j < 16; ++j) a[j] = r[kb + j];
  return a;
}

__device__ inline v16h frag_a_cm(const _Float16* p, int ld) {
  // A[m][k] = p[k*ld + m]   (transposed source)
  const int lane = threadIdx.x & 31;
  const int m  = lane & 15;
  const int kb = (lane & 16) ? 16 : 0;
  v16h a;
#pragma unroll
  for (int j = 0; j < 8; ++j) {
    const int k = kb + 2*j;
    a[2*j]   = p[(size_t)k * ld + m];
    a[2*j+1] = p[(size_t)(k+1) * ld + m];
  }
  return a;
}

// B 32x16 (KxN): lane n = lane&15, K range [0,16) lanes 0-15, [16,32) lanes 16-31
__device__ inline v16h frag_b_rm(const _Float16* p, int ld) {
  // B[k][n] = p[k*ld + n]
  const int lane = threadIdx.x & 31;
  const int n  = lane & 15;
  const int kb = (lane & 16) ? 16 : 0;
  v16h b;
#pragma unroll
  for (int j = 0; j < 16; ++j) b[j] = p[(size_t)(kb + j) * ld + n];
  return b;
}

__device__ inline v16h frag_b_tr(const _Float16* p, int ld) {
  // B[k][n] = p[n*ld + k]  (B supplied transposed, e.g. weights NxK) -> contiguous
  const int lane = threadIdx.x & 31;
  const int n  = lane & 15;
  const int kb = (lane & 16) ? 16 : 0;
  const _Float16* r = p + (size_t)n * ld + kb;
  v16h b;
#pragma unroll
  for (int j = 0; j < 16; ++j) b[j] = r[j];
  return b;
}

__device__ inline v8f wmma_f16(v16h a, v16h b, v8f c) {
  return __builtin_amdgcn_wmma_f32_16x16x32_f16(false, a, false, b, (short)0, c, false, false);
}

// ---------------------------------------------------------------------------
// elementwise f32 -> f16 convert
// ---------------------------------------------------------------------------
__global__ __launch_bounds__(256) void cvt_f32_f16(const float* __restrict__ x,
                                                   _Float16* __restrict__ y, int n) {
  int i = blockIdx.x * 256 + threadIdx.x;
  if (i < n) y[i] = (_Float16)x[i];
}

// ---------------------------------------------------------------------------
// per-token L2 normalize; emit f16 raw + f16 normalized
// ---------------------------------------------------------------------------
__global__ __launch_bounds__(256) void rownorm_prep(const float* __restrict__ src,
                                                    _Float16* __restrict__ src_h,
                                                    _Float16* __restrict__ src1_h) {
  const int row = blockIdx.x;
  const float* x = src + (size_t)row * DD;
  float ss = 0.f;
  for (int i = threadIdx.x; i < DD; i += 256) { float v = x[i]; ss += v * v; }
  __shared__ float red[256];
  red[threadIdx.x] = ss; __syncthreads();
  for (int s = 128; s > 0; s >>= 1) {
    if (threadIdx.x < s) red[threadIdx.x] += red[threadIdx.x + s];
    __syncthreads();
  }
  const float rn = rsqrtf(red[0]);
  for (int i = threadIdx.x; i < DD; i += 256) {
    float v = x[i];
    src_h [(size_t)row * DD + i] = (_Float16)v;
    src1_h[(size_t)row * DD + i] = (_Float16)(v * rn);
  }
}

// ---------------------------------------------------------------------------
// Generic WMMA GEMM:  C[M,N] = act(A[M,K] * W[N,K]^T + bias[N])  (+= if accumulate)
// 128x64 block tile, BK=32, 8 waves; each wave computes 32x32 (4 WMMAs/K-step,
// A & B fragments each reused twice -> half the LDS traffic per FLOP)
// ---------------------------------------------------------------------------
#define BM 128
#define BN 64
#define BK 32
#define LDS_S 40   // padded f16 stride (80B rows, 16B aligned)

__global__ __launch_bounds__(256)
void wmma_gemm_nt(const _Float16* __restrict__ A, const _Float16* __restrict__ W,
                  const float* __restrict__ bias,
                  float* __restrict__ Cf, _Float16* __restrict__ Ch,
                  int M, int N, int K, int act, int accumulate) {
  __shared__ _Float16 As[BM * LDS_S];
  __shared__ _Float16 Bs[BN * LDS_S];
  const int bm = blockIdx.x * BM;
  const int bn = blockIdx.y * BN;
  const int tid  = threadIdx.x;
  const int wave = tid >> 5;
  const int tm = (wave & 3) * 32;          // 4 row groups of 32
  const int tn = (wave >> 2) * 32;         // 2 col groups of 32
  // A staging: 128 rows x 4 chunks(8h) = 512 chunks, 2 per thread
  const int ac0 = tid * 2, ac1 = tid * 2 + 1;
  // B staging: 64 rows x 4 chunks = 256 chunks, 1 per thread
  const int brow = tid >> 2, bcol = (tid & 3) * 8;

  v8f acc00 = {}, acc01 = {}, acc10 = {}, acc11 = {};

  for (int k0 = 0; k0 < K; k0 += BK) {
    *(h8*)(As + (ac0 >> 2) * LDS_S + (ac0 & 3) * 8) =
        *(const h8*)(A + (size_t)(bm + (ac0 >> 2)) * K + k0 + (ac0 & 3) * 8);
    *(h8*)(As + (ac1 >> 2) * LDS_S + (ac1 & 3) * 8) =
        *(const h8*)(A + (size_t)(bm + (ac1 >> 2)) * K + k0 + (ac1 & 3) * 8);
    *(h8*)(Bs + brow * LDS_S + bcol) =
        *(const h8*)(W + (size_t)(bn + brow) * K + k0 + bcol);
    if (k0 + 2 * BK <= K - BK) {
      __builtin_prefetch(A + (size_t)(bm + (ac0 >> 2)) * K + k0 + 2 * BK + (ac0 & 3) * 8, 0, 1);
      __builtin_prefetch(W + (size_t)(bn + brow) * K + k0 + 2 * BK + bcol, 0, 1);
    }
    __syncthreads();
    v16h a0 = frag_a_rm(As + tm * LDS_S, LDS_S);
    v16h a1 = frag_a_rm(As + (tm + 16) * LDS_S, LDS_S);
    v16h b0 = frag_b_tr(Bs + tn * LDS_S, LDS_S);
    v16h b1 = frag_b_tr(Bs + (tn + 16) * LDS_S, LDS_S);
    acc00 = wmma_f16(a0, b0, acc00);
    acc01 = wmma_f16(a0, b1, acc01);
    acc10 = wmma_f16(a1, b0, acc10);
    acc11 = wmma_f16(a1, b1, acc11);
    __syncthreads();
  }

  const int lane = tid & 31;
  const int cn = lane & 15;
  const int mb = (lane & 16) ? 8 : 0;
#pragma unroll
  for (int ti = 0; ti < 2; ++ti) {
#pragma unroll
    for (int tj = 0; tj < 2; ++tj) {
      const v8f acc = ti ? (tj ? acc11 : acc10) : (tj ? acc01 : acc00);
      const int ncol = bn + tn + 16 * tj + cn;
      const float bv = bias ? bias[ncol] : 0.f;
#pragma unroll
      for (int r = 0; r < 8; ++r) {
        const int row = bm + tm + 16 * ti + mb + r;
        float x = acc[r] + bv;
        if (act == 1) x = 0.5f * x * (1.f + erff(x * 0.70710678f)); // exact GELU
        const size_t o = (size_t)row * N + ncol;
        if (Cf) { if (accumulate) Cf[o] += x; else Cf[o] = x; }
        if (Ch) Ch[o] = (_Float16)x;
      }
    }
  }
}

// ---------------------------------------------------------------------------
// FFT-decay filter as first-order recurrence:
//   q1[t] = q1[t-1]/r + a*r^(H-1)*q[t],  a = sigmoid(alpha[d]), r = 1-a
// ---------------------------------------------------------------------------
__global__ __launch_bounds__(256)
void decay_scan(const float* __restrict__ qk, const float* __restrict__ alpha,
                _Float16* __restrict__ q1h, _Float16* __restrict__ k1h) {
  const int idx = blockIdx.x * 256 + threadIdx.x;   // 65536
  const int d  = idx & 63;
  const int nh = (idx >> 6) & 7;
  const int bn = idx >> 9;
  const float a = 1.f / (1.f + __expf(-alpha[d]));
  const float r = 1.f - a;
  const float invr = 1.f / r;
  const float c2 = a * __powf(r, (float)(HH - 1));
  const float* q = qk + (size_t)bn * HH * 1024 + nh * 64 + d;
  const float* k = q + 512;
  _Float16* q1 = q1h + ((size_t)(bn * NHD + nh) * HH) * HD + d;
  _Float16* k1 = k1h + ((size_t)(bn * NHD + nh) * HH) * HD + d;
  float aq = 0.f, ak = 0.f;
  for (int t = 0; t < HH; ++t) {
    aq = aq * invr + c2 * q[(size_t)t * 1024];
    ak = ak * invr + c2 * k[(size_t)t * 1024];
    q1[(size_t)t * HD] = (_Float16)aq;
    k1[(size_t)t * HD] = (_Float16)ak;
  }
}

// ---------------------------------------------------------------------------
// Token attention, pass 1: P = softmax(q1 * k1^T * sqrt(HD))  per slice
// ---------------------------------------------------------------------------
__global__ __launch_bounds__(32)
void tok_scores(const _Float16* __restrict__ q1, const _Float16* __restrict__ k1,
                _Float16* __restrict__ P) {
  const int strip = blockIdx.x;            // 0..11
  const int slice = blockIdx.y;            // 0..1023
  const _Float16* qs = q1 + (size_t)slice * HH * HD + (size_t)strip * 16 * HD;
  const _Float16* ks = k1 + (size_t)slice * HH * HD;
  v8f acc[12];
#pragma unroll
  for (int nt = 0; nt < 12; ++nt) acc[nt] = (v8f){};
#pragma unroll
  for (int nt = 0; nt < 12; ++nt)
#pragma unroll
    for (int kk = 0; kk < HD; kk += 32) {
      v16h a = frag_a_rm(qs + kk, HD);
      v16h b = frag_b_tr(ks + (size_t)nt * 16 * HD + kk, HD);   // B = k1^T
      acc[nt] = wmma_f16(a, b, acc[nt]);
    }
  const int lane = threadIdx.x & 31;
  const int cn = lane & 15;
  const int mb = (lane & 16) ? 8 : 0;
  const float scale = 8.0f;                // sqrt(HD)
#pragma unroll
  for (int r = 0; r < 8; ++r) {
    float mx = -3.0e38f;
#pragma unroll
    for (int nt = 0; nt < 12; ++nt) { acc[nt][r] *= scale; mx = fmaxf(mx, acc[nt][r]); }
#pragma unroll
    for (int m = 1; m < 16; m <<= 1) mx = fmaxf(mx, __shfl_xor(mx, m, 32));
    float sum = 0.f;
#pragma unroll
    for (int nt = 0; nt < 12; ++nt) { float e = __expf(acc[nt][r] - mx); acc[nt][r] = e; sum += e; }
#pragma unroll
    for (int m = 1; m < 16; m <<= 1) sum += __shfl_xor(sum, m, 32);
    const float inv = 1.f / sum;
    const int row = strip * 16 + mb + r;
#pragma unroll
    for (int nt = 0; nt < 12; ++nt)
      P[(size_t)slice * HH * HH + (size_t)row * HH + nt * 16 + cn] = (_Float16)(acc[nt][r] * inv);
  }
}

// ---------------------------------------------------------------------------
// Token attention, pass 2: out_tok = P(192x192) @ V(192x64)
// V tile (192x64 f16, row stride 512) staged into LDS by the Tensor Data Mover
// ---------------------------------------------------------------------------
__global__ __launch_bounds__(384)
void tok_av(const _Float16* __restrict__ P, const _Float16* __restrict__ vh,
            float* __restrict__ out) {
  const int slice = blockIdx.x;
  const int bn = slice >> 3, nh = slice & 7;
  __shared__ _Float16 vs[HH * HD];
  const _Float16* vsrc = vh + ((size_t)(bn * HH)) * DD + nh * 64;
#if __has_builtin(__builtin_amdgcn_tensor_load_to_lds)
  if (threadIdx.x < 32) {  // one wave issues the TDM descriptor (EXEC ignored)
    const unsigned lds_addr = (unsigned)(size_t)(void*)vs;
    const unsigned long long ga = (unsigned long long)(size_t)vsrc;
    u32x4 g0;
    g0[0] = 1u;                                             // count=1, user mode
    g0[1] = lds_addr;                                       // LDS dest (bytes)
    g0[2] = (unsigned)(ga & 0xffffffffu);                   // global_addr[31:0]
    g0[3] = (unsigned)((ga >> 32) & 0x1ffffffu) | (2u << 30); // addr[56:32], type=2
    i32x8 g1 = (i32x8){};
    g1[0] = 1 << 16;                                        // data_size=2 bytes
    g1[1] = (HD & 0xffff) << 16;                            // tensor_dim0 = 64
    g1[2] = (HH & 0xffff) << 16;                            // tensor_dim1 = 192
    g1[3] = HD << 16;                                       // tile_dim0 = 64
    g1[4] = HH;                                             // tile_dim1 = 192
    g1[5] = DD;                                             // tensor_dim0_stride = 512
    i32x4 gz = (i32x4){};
#if defined(__clang_major__) && (__clang_major__ >= 23)
    i32x8 gz8 = (i32x8){};
    __builtin_amdgcn_tensor_load_to_lds(g0, g1, gz, gz, gz8, 0);
#else
    __builtin_amdgcn_tensor_load_to_lds(g0, g1, gz, gz, 0);
#endif
    __builtin_amdgcn_s_wait_tensorcnt(0);
  }
#else
  for (int i = threadIdx.x; i < HH * HD / 8; i += 384) {
    const int a = (i * 8) / HD, c = (i * 8) % HD;
    *(h8*)(vs + a * HD + c) = *(const h8*)(vsrc + (size_t)a * DD + c);
  }
#endif
  __syncthreads();
  const int wave = threadIdx.x >> 5;       // strip 0..11
  const _Float16* ps = P + (size_t)slice * HH * HH + (size_t)wave * 16 * HH;
  const int lane = threadIdx.x & 31;
  const int cn = lane & 15;
  const int mb = (lane & 16) ? 8 : 0;
#pragma unroll
  for (int nt = 0; nt < 4; ++nt) {
    v8f acc = {};
#pragma unroll
    for (int kk = 0; kk < HH; kk += 32) {
      v16h a = frag_a_rm(ps + kk, HH);
      v16h b = frag_b_rm(vs + (size_t)kk * HD + nt * 16, HD);
      acc = wmma_f16(a, b, acc);
    }
#pragma unroll
    for (int r = 0; r < 8; ++r) {
      const int row = wave * 16 + mb + r;
      out[((size_t)(bn * HH + row)) * DD + nh * 64 + nt * 16 + cn] = acc[r];
    }
  }
}

// ---------------------------------------------------------------------------
// Hidden attention, pass 1: Ah = softmax(q^T k * sqrt(H))  (64x64 per slice)
// ---------------------------------------------------------------------------
__global__ __launch_bounds__(128)
void hid_scores(const _Float16* __restrict__ qkh, _Float16* __restrict__ Ah) {
  const int slice = blockIdx.x;
  const int bn = slice >> 3, nh = slice & 7;
  const int strip = threadIdx.x >> 5;      // 0..3  (e rows)
  const _Float16* qb = qkh + ((size_t)(bn * HH)) * 1024 + nh * 64;   // q[a][e]
  const _Float16* kb = qb + 512;                                     // k[a][f]
  v8f acc[4];
#pragma unroll
  for (int nt = 0; nt < 4; ++nt) acc[nt] = (v8f){};
#pragma unroll
  for (int nt = 0; nt < 4; ++nt)
    for (int kk = 0; kk < HH; kk += 32) {
      v16h a = frag_a_cm(qb + (size_t)kk * 1024 + strip * 16, 1024); // A = q^T
      v16h b = frag_b_rm(kb + (size_t)kk * 1024 + nt * 16, 1024);    // B = k
      acc[nt] = wmma_f16(a, b, acc[nt]);
    }
  const int lane = threadIdx.x & 31;
  const int cn = lane & 15;
  const int mb = (lane & 16) ? 8 : 0;
  const float scale = 13.8564065f;         // sqrt(192)
#pragma unroll
  for (int r = 0; r < 8; ++r) {
    float mx = -3.0e38f;
#pragma unroll
    for (int nt = 0; nt < 4; ++nt) { acc[nt][r] *= scale; mx = fmaxf(mx, acc[nt][r]); }
#pragma unroll
    for (int m = 1; m < 16; m <<= 1) mx = fmaxf(mx, __shfl_xor(mx, m, 32));
    float sum = 0.f;
#pragma unroll
    for (int nt = 0; nt < 4; ++nt) { float e = __expf(acc[nt][r] - mx); acc[nt][r] = e; sum += e; }
#pragma unroll
    for (int m = 1; m < 16; m <<= 1) sum += __shfl_xor(sum, m, 32);
    const float inv = 1.f / sum;
    const int row = strip * 16 + mb + r;
#pragma unroll
    for (int nt = 0; nt < 4; ++nt)
      Ah[(size_t)slice * HD * HD + (size_t)row * HD + nt * 16 + cn] = (_Float16)(acc[nt][r] * inv);
  }
}

// ---------------------------------------------------------------------------
// Hidden attention, pass 2: out_hid = V(192x64) @ Ah^T(64x64), token-major f32
// ---------------------------------------------------------------------------
__global__ __launch_bounds__(384)
void hid_av(const _Float16* __restrict__ Ah, const _Float16* __restrict__ vh,
            float* __restrict__ out) {
  const int slice = blockIdx.x;
  const int bn = slice >> 3, nh = slice & 7;
  const int wave = threadIdx.x >> 5;       // strip 0..11 (token rows)
  const _Float16* vb = vh + ((size_t)(bn * HH + wave * 16)) * DD + nh * 64;
  const _Float16* ab = Ah + (size_t)slice * HD * HD;
  const int lane = threadIdx.x & 31;
  const int cn = lane & 15;
  const int mb = (lane & 16) ? 8 : 0;
#pragma unroll
  for (int nt = 0; nt < 4; ++nt) {
    v8f acc = {};
#pragma unroll
    for (int kk = 0; kk < HD; kk += 32) {
      v16h a = frag_a_rm(vb + kk, DD);                 // A = v[a][f]
      v16h b = frag_b_tr(ab + (size_t)nt * 16 * HD + kk, HD); // B[f][e] = Ah[e][f]
      acc = wmma_f16(a, b, acc);
    }
#pragma unroll
    for (int r = 0; r < 8; ++r) {
      const int row = wave * 16 + mb + r;
      out[((size_t)(bn * HH + row)) * DD + nh * 64 + nt * 16 + cn] = acc[r];
    }
  }
}

// ---------------------------------------------------------------------------
// BatchNorm: per-channel stats over all rows (one block per channel)
// ---------------------------------------------------------------------------
__global__ __launch_bounds__(256)
void bn_stats(const float* __restrict__ x, const float* __restrict__ x2,
              float* __restrict__ mean, float* __restrict__ var, int rows) {
  const int c = blockIdx.x;
  float s = 0.f, ss = 0.f;
  for (int r = threadIdx.x; r < rows; r += 256) {
    float v = x[(size_t)r * DD + c];
    if (x2) v += x2[(size_t)r * DD + c];
    s += v; ss += v * v;
  }
  __shared__ float rs[256], rss[256];
  rs[threadIdx.x] = s; rss[threadIdx.x] = ss; __syncthreads();
  for (int k = 128; k > 0; k >>= 1) {
    if (threadIdx.x < k) { rs[threadIdx.x] += rs[threadIdx.x + k]; rss[threadIdx.x] += rss[threadIdx.x + k]; }
    __syncthreads();
  }
  if (threadIdx.x == 0) {
    const float m = rs[0] / rows;
    mean[c] = m;
    var[c] = rss[0] / rows - m * m;
  }
}

__global__ __launch_bounds__(256)
void bn_apply(const float* __restrict__ x, const float* __restrict__ x2,
              const float* __restrict__ mean, const float* __restrict__ var,
              const float* __restrict__ g, const float* __restrict__ b,
              float* __restrict__ yf, _Float16* __restrict__ yh, int rows) {
  const size_t i = (size_t)blockIdx.x * 256 + threadIdx.x;
  if (i >= (size_t)rows * DD) return;
  const int c = (int)(i % DD);
  float v = x[i];
  if (x2) v += x2[i];
  const float y = (v - mean[c]) * rsqrtf(var[c] + EPS) * g[c] + b[c];
  if (yf) yf[i] = y;
  if (yh) yh[i] = (_Float16)y;
}

// ---------------------------------------------------------------------------
// Host orchestration
// ---------------------------------------------------------------------------
extern "C" void kernel_launch(void* const* d_in, const int* in_sizes, int n_in,
                              void* d_out, int out_size, void* d_ws, size_t ws_size,
                              hipStream_t stream) {
  const float* src    = (const float*)d_in[0];
  const float* qk_w   = (const float*)d_in[1];
  const float* qk_b   = (const float*)d_in[2];
  const float* v_w    = (const float*)d_in[3];
  const float* v_b    = (const float*)d_in[4];
  const float* alpha  = (const float*)d_in[5];
  const float* g_pre1 = (const float*)d_in[6];
  const float* b_pre1 = (const float*)d_in[7];
  const float* g_pre2 = (const float*)d_in[8];
  const float* b_pre2 = (const float*)d_in[9];
  const float* f1w1   = (const float*)d_in[10];
  const float* f1b1   = (const float*)d_in[11];
  const float* f1w2   = (const float*)d_in[12];
  const float* f1b2   = (const float*)d_in[13];
  const float* f2w1   = (const float*)d_in[14];
  const float* f2b1   = (const float*)d_in[15];
  const float* f2w2   = (const float*)d_in[16];
  const float* f2b2   = (const float*)d_in[17];
  const float* g_attn = (const float*)d_in[18];
  const float* b_attn = (const float*)d_in[19];

  char* base = (char*)d_ws;
  size_t off = 0;
  auto alloc = [&](size_t bytes) -> void* {
    off = (off + 255) & ~(size_t)255;
    void* p = base + off;
    off += bytes;
    return p;
  };

  _Float16* src_h   = (_Float16*)alloc((size_t)TOK * DD * 2);
  _Float16* src1_h  = (_Float16*)alloc((size_t)TOK * DD * 2);
  _Float16* qkw_h   = (_Float16*)alloc((size_t)2 * DD * DD * 2);
  _Float16* vw_h    = (_Float16*)alloc((size_t)DD * DD * 2);
  _Float16* f1w1_h  = (_Float16*)alloc((size_t)DFF * DD * 2);
  _Float16* f1w2_h  = (_Float16*)alloc((size_t)DD * DFF * 2);
  _Float16* f2w1_h  = (_Float16*)alloc((size_t)DFF * DD * 2);
  _Float16* f2w2_h  = (_Float16*)alloc((size_t)DD * DFF * 2);
  float*    qk_f    = (float*)   alloc((size_t)TOK * 1024 * 4);
  _Float16* qk_h    = (_Float16*)alloc((size_t)TOK * 1024 * 2);
  _Float16* v_h     = (_Float16*)alloc((size_t)TOK * DD * 2);
  _Float16* q1_h    = (_Float16*)alloc((size_t)NSLICE * HH * HD * 2);
  _Float16* k1_h    = (_Float16*)alloc((size_t)NSLICE * HH * HD * 2);
  _Float16* P_h     = (_Float16*)alloc((size_t)NSLICE * HH * HH * 2);
  _Float16* A_h     = (_Float16*)alloc((size_t)NSLICE * HD * HD * 2);
  float*    out_tok = (float*)   alloc((size_t)TOK * DD * 4);
  float*    out_hid = (float*)   alloc((size_t)TOK * DD * 4);
  _Float16* o3_h    = (_Float16*)alloc((size_t)TOK * DD * 2);
  _Float16* o2_h    = (_Float16*)alloc((size_t)TOK * DD * 2);
  _Float16* hbuf    = (_Float16*)alloc((size_t)TOK * DFF * 2);
  float*    src2    = (float*)   alloc((size_t)TOK * DD * 4);
  float*    stats   = (float*)   alloc((size_t)6 * DD * 4);
  float* mean3 = stats;            float* var3 = stats + DD;
  float* mean1 = stats + 2 * DD;   float* var1 = stats + 3 * DD;
  float* meanF = stats + 4 * DD;   float* varF = stats + 5 * DD;
  (void)ws_size; (void)n_in; (void)in_sizes; (void)out_size;

  auto cvt = [&](const float* s, _Float16* d, int n) {
    cvt_f32_f16<<<(n + 255) / 256, 256, 0, stream>>>(s, d, n);
  };
  cvt(qk_w, qkw_h, 2 * DD * DD);
  cvt(v_w,  vw_h,  DD * DD);
  cvt(f1w1, f1w1_h, DFF * DD);
  cvt(f1w2, f1w2_h, DD * DFF);
  cvt(f2w1, f2w1_h, DFF * DD);
  cvt(f2w2, f2w2_h, DD * DFF);

  rownorm_prep<<<TOK, 256, 0, stream>>>(src, src_h, src1_h);

  // qk = src1 @ qk_w^T + qk_b  (f32 for decay scan, f16 for hidden attention)
  wmma_gemm_nt<<<dim3(TOK / BM, 1024 / BN), 256, 0, stream>>>(
      src1_h, qkw_h, qk_b, qk_f, qk_h, TOK, 1024, DD, 0, 0);
  // v = src @ v_w^T + v_b (f16 only)
  wmma_gemm_nt<<<dim3(TOK / BM, DD / BN), 256, 0, stream>>>(
      src_h, vw_h, v_b, (float*)nullptr, v_h, TOK, DD, DD, 0, 0);

  // decay filter along token axis -> q1, k1 (slice-major f16)
  decay_scan<<<(NSL * NHD * HD) / 256, 256, 0, stream>>>(qk_f, alpha, q1_h, k1_h);

  // token-axis attention
  tok_scores<<<dim3(HH / 16, NSLICE), 32, 0, stream>>>(q1_h, k1_h, P_h);
  tok_av<<<NSLICE, 384, 0, stream>>>(P_h, v_h, out_tok);

  // hidden-axis attention
  hid_scores<<<NSLICE, 128, 0, stream>>>(qk_h, A_h);
  hid_av<<<NSLICE, 384, 0, stream>>>(A_h, v_h, out_hid);

  // o3 = BN(out_tok; g_pre2,b_pre2), o2 = BN(out_hid; g_pre1,b_pre1) -> f16
  const int ELT_BLK = (int)(((size_t)TOK * DD + 255) / 256);
  bn_stats<<<DD, 256, 0, stream>>>(out_tok, nullptr, mean3, var3, TOK);
  bn_apply<<<ELT_BLK, 256, 0, stream>>>(out_tok, nullptr, mean3, var3, g_pre2, b_pre2,
                                        (float*)nullptr, o3_h, TOK);
  bn_stats<<<DD, 256, 0, stream>>>(out_hid, nullptr, mean1, var1, TOK);
  bn_apply<<<ELT_BLK, 256, 0, stream>>>(out_hid, nullptr, mean1, var1, g_pre1, b_pre1,
                                        (float*)nullptr, o2_h, TOK);

  // FF branch 1: src2 = (gelu(o3 @ f1w1^T + b1)) @ f1w2^T + b2
  wmma_gemm_nt<<<dim3(TOK / BM, DFF / BN), 256, 0, stream>>>(
      o3_h, f1w1_h, f1b1, (float*)nullptr, hbuf, TOK, DFF, DD, 1, 0);
  wmma_gemm_nt<<<dim3(TOK / BM, DD / BN), 256, 0, stream>>>(
      hbuf, f1w2_h, f1b2, src2, (_Float16*)nullptr, TOK, DD, DFF, 0, 0);
  // FF branch 2 accumulates into src2
  wmma_gemm_nt<<<dim3(TOK / BM, DFF / BN), 256, 0, stream>>>(
      o2_h, f2w1_h, f2b1, (float*)nullptr, hbuf, TOK, DFF, DD, 1, 0);
  wmma_gemm_nt<<<dim3(TOK / BM, DD / BN), 256, 0, stream>>>(
      hbuf, f2w2_h, f2b2, src2, (_Float16*)nullptr, TOK, DD, DFF, 0, 1);

  // out = BN(src + src2; g_attn, b_attn)
  bn_stats<<<DD, 256, 0, stream>>>(src, src2, meanF, varF, TOK);
  bn_apply<<<ELT_BLK, 256, 0, stream>>>(src, src2, meanF, varF, g_attn, b_attn,
                                        (float*)d_out, (_Float16*)nullptr, TOK);
}
